// DeformableTransformerDecoderLayer_61641370632780
// MI455X (gfx1250) — matrix operand order repaired and checked by reading, compiled, not run on hardware
//
#include <hip/hip_runtime.h>
#include <hip/hip_bf16.h>
#include <math.h>

#define BB 16
#define NQ 300
#define CC 256
#define HH 8
#define LL 4
#define PP 4
#define DFFN 1024
#define DH 32
#define LEN_IN 19947           // 100*150 + 50*75 + 25*38 + 13*19
#define ROWS (BB*NQ)           // 4800
#define MV (BB*LEN_IN)         // 319152

typedef __attribute__((ext_vector_type(16))) _Float16 v16h;
typedef __attribute__((ext_vector_type(8)))  _Float16 v8h;
typedef __attribute__((ext_vector_type(8)))  float    v8f;
typedef __attribute__((ext_vector_type(4)))  float    v4f;
typedef __attribute__((ext_vector_type(4)))  int      v4i;

// CDNA5 async global->LDS path (ASYNCcnt), guarded so compile never breaks.
#if defined(__has_builtin)
#if __has_builtin(__builtin_amdgcn_global_load_async_to_lds_b128) && \
    __has_builtin(__builtin_amdgcn_s_wait_asynccnt)
#define ASYNC_LDS 1
#endif
#endif

#ifdef ASYNC_LDS
typedef __attribute__((address_space(1))) v4i* gas_v4i;   // global (AS1) int4*
typedef __attribute__((address_space(3))) v4i* las_v4i;   // LDS (AS3) int4*
__device__ inline void async_copy16(const void* gsrc, void* ldst) {
    v4i* gp = (v4i*)(uintptr_t)gsrc;   // strip const, reinterpret as int4*
    v4i* lp = (v4i*)ldst;
    __builtin_amdgcn_global_load_async_to_lds_b128((gas_v4i)gp, (las_v4i)lp, 0, 0);
}
#endif

// ---------------- wave32 reductions ----------------
__device__ inline float wave_sum(float v) {
#pragma unroll
    for (int i = 16; i > 0; i >>= 1) v += __shfl_xor(v, i);
    return v;
}
__device__ inline float wave_max(float v) {
#pragma unroll
    for (int i = 16; i > 0; i >>= 1) v = fmaxf(v, __shfl_xor(v, i));
    return v;
}

// ---------------- fp32 -> f16 weight conversion ----------------
__global__ void cvt_f32_f16_kernel(const float* __restrict__ src, _Float16* __restrict__ dst, int n) {
    int i = blockIdx.x * blockDim.x + threadIdx.x;
    if (i < n) dst[i] = (_Float16)src[i];
}

// ---------------- elementwise add ----------------
__global__ void add_kernel(const float* __restrict__ a, const float* __restrict__ b,
                           float* __restrict__ o, int n) {
    int i = blockIdx.x * blockDim.x + threadIdx.x;
    if (i < n) o[i] = a[i] + b[i];
}

// ---------------- generic WMMA GEMM: Y[M,N] = X[M,K] @ W[N,K]^T + bias ----------------
// X fp32 (converted to f16 through LDS), W pre-converted f16 row-major [N,K].
// Block tile 64M x 128N, K-step 32, double-buffered LDS (one barrier/iter).
// 8 waves, each owns a 32x32 quadrant (2x2 wmma tiles).
template <bool OUT_F16>
__global__ __launch_bounds__(256)
void gemm_xw_kernel(const float* __restrict__ X, const _Float16* __restrict__ W,
                    const float* __restrict__ bias, const unsigned char* __restrict__ rowMask,
                    void* __restrict__ Yv, int M, int N, int K, int relu)
{
    __shared__ __attribute__((aligned(32))) _Float16 ldsA[2][64 * 32];
    __shared__ __attribute__((aligned(32))) _Float16 ldsB[2][128 * 32];

    const int tid  = threadIdx.x;
    const int lane = tid & 31;
    const int wave = tid >> 5;
    const int hi   = lane >> 4;     // which 16-lane half
    const int r    = lane & 15;

    const int blockN = blockIdx.x * 128;
    const int blockM = blockIdx.y * 64;

    const int wm = wave & 1;        // M quadrant (0/32)
    const int wn = wave >> 1;       // N quadrant (0/32/64/96)

    v8f acc[2][2] = {};

    // A staging: row = tid>>2 (0..63), 8 fp32 at col (tid&3)*8
    const int arow = tid >> 2;
    const int acol = (tid & 3) * 8;
    // B staging: n = tid>>1 (0..127), 16 halves at k (tid&1)*16
    const int brow = tid >> 1;
    const int bk   = (tid & 1) * 16;

    int gr = blockM + arow;
    if (gr >= M) gr = M - 1;        // clamp: padded rows never stored

    const float*    apBase = X + (size_t)gr * K + acol;
    const _Float16* bpBase = W + (size_t)(blockN + brow) * K + bk;
    const int aslot = arow * 32 + acol;
    const int bslot = brow * 32 + bk;

    // ---- helpers ----
    auto stage_b = [&](const _Float16* bp, _Float16* dst) {
#ifdef ASYNC_LDS
        async_copy16(bp, dst);
        async_copy16(bp + 8, dst + 8);
#else
        *(v16h*)dst = *(const v16h*)bp;
#endif
    };
    auto store_a = [&](v4f a0, v4f a1, _Float16* dA) {
        dA[0] = (_Float16)a0[0]; dA[1] = (_Float16)a0[1];
        dA[2] = (_Float16)a0[2]; dA[3] = (_Float16)a0[3];
        dA[4] = (_Float16)a1[0]; dA[5] = (_Float16)a1[1];
        dA[6] = (_Float16)a1[2]; dA[7] = (_Float16)a1[3];
    };

    // ---- prologue: stage tile 0 into buffer 0 ----
    {
        v4f a0 = *(const v4f*)apBase;
        v4f a1 = *(const v4f*)(apBase + 4);
        store_a(a0, a1, &ldsA[0][aslot]);
        stage_b(bpBase, &ldsB[0][bslot]);
    }
#ifdef ASYNC_LDS
    __builtin_amdgcn_s_wait_asynccnt(0);
#endif
    __syncthreads();

    int buf = 0;
    for (int k0 = 0; k0 < K; k0 += 32, buf ^= 1) {
        const int nk = k0 + 32;
        const bool have_next = nk < K;
        v4f a0, a1;
        if (have_next) {
            // issue next-tile loads while this tile computes
            a0 = *(const v4f*)(apBase + nk);
            a1 = *(const v4f*)(apBase + nk + 4);
            stage_b(bpBase + nk, &ldsB[buf ^ 1][bslot]);
            __builtin_prefetch(apBase + nk + 32, 0, 1);
        }

        // ---- fragments from current buffer ----
        // A 16x32, this code's contiguous k order (same as B): lanes0-15 K0..15, lanes16-31 K16..31, row = r
        v16h afrag[2];
#pragma unroll
        for (int i = 0; i < 2; ++i) {
            int row = wm * 32 + i * 16 + r;
            afrag[i] = *(const v16h*)&ldsA[buf][row * 32 + hi * 16];
        }
        // B 32x16: lanes0-15 K0..15, lanes16-31 K16..31, col = r
        v16h bfrag[2];
#pragma unroll
        for (int j = 0; j < 2; ++j) {
            int col = wn * 32 + j * 16 + r;
            bfrag[j] = *(const v16h*)&ldsB[buf][col * 32 + hi * 16];
        }
#pragma unroll
        for (int i = 0; i < 2; ++i)
#pragma unroll
            for (int j = 0; j < 2; ++j)
                acc[i][j] = __builtin_amdgcn_wmma_f32_16x16x32_f16(
                    false, afrag[i], false, bfrag[j],
                    (short)0, acc[i][j], false, false);

        if (have_next) {
            store_a(a0, a1, &ldsA[buf ^ 1][aslot]);
#ifdef ASYNC_LDS
            __builtin_amdgcn_s_wait_asynccnt(0);
#endif
            __syncthreads();
        }
    }

    // ---- epilogue: C/D layout: VGPR v -> M = hi*8 + v, N = r ----
#pragma unroll
    for (int i = 0; i < 2; ++i) {
#pragma unroll
        for (int j = 0; j < 2; ++j) {
            int colg = blockN + wn * 32 + j * 16 + r;
            float bv = bias ? bias[colg] : 0.0f;
#pragma unroll
            for (int v = 0; v < 8; ++v) {
                int rowg = blockM + wm * 32 + i * 16 + hi * 8 + v;
                if (rowg < M) {
                    float val = acc[i][j][v] + bv;
                    if (relu) val = fmaxf(val, 0.0f);
                    if (rowMask && rowMask[rowg]) val = 0.0f;
                    if (OUT_F16)
                        ((_Float16*)Yv)[(size_t)rowg * N + colg] = (_Float16)val;
                    else
                        ((float*)Yv)[(size_t)rowg * N + colg] = val;
                }
            }
        }
    }
}

// ---------------- self-attention (one wave per (b,h,q), 300 keys) ----------------
__global__ __launch_bounds__(256)
void attn_kernel(const float* __restrict__ qk, const float* __restrict__ vp,
                 float* __restrict__ out)
{
    __shared__ float s_sc[8][NQ];
    __shared__ float s_q[8][DH];
    const int wave = threadIdx.x >> 5, lane = threadIdx.x & 31;
    const int gid = blockIdx.x * 8 + wave;                  // (b*H + h)*NQ + q
    const int b = gid / (HH * NQ);
    const int rem = gid - b * HH * NQ;
    const int h = rem / NQ;
    const int qi = rem - h * NQ;
    const int qrow = b * NQ + qi;

    s_q[wave][lane] = qk[(size_t)qrow * 512 + h * DH + lane];
    __syncthreads();

    const float scale = 0.17677669529663687f;               // 1/sqrt(32)
    for (int kk = lane; kk < NQ; kk += 32) {
        const float* kp = qk + (size_t)(b * NQ + kk) * 512 + 256 + h * DH;
        float d = 0.0f;
#pragma unroll
        for (int c = 0; c < DH; ++c) d += s_q[wave][c] * kp[c];
        s_sc[wave][kk] = d * scale;
    }
    __syncthreads();

    float m = -1e30f;
    for (int kk = lane; kk < NQ; kk += 32) m = fmaxf(m, s_sc[wave][kk]);
    m = wave_max(m);
    float s = 0.0f;
    for (int kk = lane; kk < NQ; kk += 32) {
        float e = __expf(s_sc[wave][kk] - m);
        s_sc[wave][kk] = e;
        s += e;
    }
    s = wave_sum(s);
    __syncthreads();

    const float inv = 1.0f / s;
    float acc = 0.0f;
    for (int kk = 0; kk < NQ; ++kk)
        acc += s_sc[wave][kk] * vp[(size_t)(b * NQ + kk) * CC + h * DH + lane];
    out[(size_t)qrow * CC + h * DH + lane] = acc * inv;
}

// ---------------- residual + LayerNorm (one wave per row of 256) ----------------
__global__ __launch_bounds__(256)
void add_ln_kernel(const float* __restrict__ a, const float* __restrict__ rsd,
                   const float* __restrict__ g, const float* __restrict__ be,
                   float* __restrict__ out)
{
    const int wave = threadIdx.x >> 5, lane = threadIdx.x & 31;
    const int row = blockIdx.x * 8 + wave;
    const float* pa = a + (size_t)row * CC;
    const float* pr = rsd + (size_t)row * CC;
    float x[8];
    float s = 0.0f;
#pragma unroll
    for (int i = 0; i < 8; ++i) {
        int c = lane + i * 32;
        x[i] = pa[c] + pr[c];
        s += x[i];
    }
    s = wave_sum(s);
    const float mean = s * (1.0f / CC);
    float v = 0.0f;
#pragma unroll
    for (int i = 0; i < 8; ++i) { float d = x[i] - mean; v += d * d; }
    v = wave_sum(v);
    const float rstd = rsqrtf(v * (1.0f / CC) + 1e-5f);
    float* po = out + (size_t)row * CC;
#pragma unroll
    for (int i = 0; i < 8; ++i) {
        int c = lane + i * 32;
        po[c] = (x[i] - mean) * rstd * g[c] + be[c];
    }
}

// ---------------- sampling-location + attention-weight softmax prep ----------------
__global__ void prep_kernel(const float* __restrict__ off, const float* __restrict__ awlin,
                            const float* __restrict__ ref, const long long* __restrict__ sp,
                            float* __restrict__ loc, float* __restrict__ aw)
{
    int idx = blockIdx.x * blockDim.x + threadIdx.x;      // (row, h)
    if (idx >= ROWS * HH) return;
    const int row = idx >> 3, h = idx & 7;

    const float* al = awlin + (size_t)row * 128 + h * 16;
    float m = -1e30f;
#pragma unroll
    for (int i = 0; i < 16; ++i) m = fmaxf(m, al[i]);
    float e[16], s = 0.0f;
#pragma unroll
    for (int i = 0; i < 16; ++i) { e[i] = __expf(al[i] - m); s += e[i]; }
    const float inv = 1.0f / s;
    float* ao = aw + (size_t)row * 128 + h * 16;
#pragma unroll
    for (int i = 0; i < 16; ++i) ao[i] = e[i] * inv;

#pragma unroll
    for (int l = 0; l < LL; ++l) {
        const float Wl = (float)sp[2 * l + 1], Hl = (float)sp[2 * l];
        const float rx = ref[((size_t)row * LL + l) * 2 + 0];
        const float ry = ref[((size_t)row * LL + l) * 2 + 1];
#pragma unroll
        for (int p = 0; p < PP; ++p) {
            int col = h * 32 + l * 8 + p * 2;
            loc[(size_t)row * 256 + col]     = rx + off[(size_t)row * 256 + col]     / Wl;
            loc[(size_t)row * 256 + col + 1] = ry + off[(size_t)row * 256 + col + 1] / Hl;
        }
    }
}

// ---------------- deformable bilinear sampling (one wave per (b,q,h), lane=channel) ----------------
__device__ inline float fetch_val(const _Float16* __restrict__ v, size_t vbase,
                                  long long st, int x, int y, int Wl, int Hl)
{
    if (x < 0 || x >= Wl || y < 0 || y >= Hl) return 0.0f;
    return (float)v[vbase + (size_t)(st + (long long)y * Wl + x) * CC];
}

__global__ __launch_bounds__(256)
void sample_kernel(const _Float16* __restrict__ value, const float* __restrict__ loc,
                   const float* __restrict__ aw, const long long* __restrict__ sp,
                   const long long* __restrict__ ls, float* __restrict__ out)
{
    const int wave = threadIdx.x >> 5, lane = threadIdx.x & 31;
    const int gid = blockIdx.x * 8 + wave;                 // (b*NQ + q)*H + h
    const int b = gid / (NQ * HH);
    const int rem = gid - b * NQ * HH;
    const int qi = rem / HH;
    const int h = rem - qi * HH;
    const int row = b * NQ + qi;

    const size_t vbase = ((size_t)b * LEN_IN) * CC + h * DH + lane;
    float acc = 0.0f;
#pragma unroll
    for (int l = 0; l < LL; ++l) {
        const int Hl = (int)sp[2 * l], Wl = (int)sp[2 * l + 1];
        const long long st = ls[l];
#pragma unroll
        for (int p = 0; p < PP; ++p) {
            const int col = h * 32 + l * 8 + p * 2;
            const float lx = loc[(size_t)row * 256 + col];
            const float ly = loc[(size_t)row * 256 + col + 1];
            const float w = aw[(size_t)row * 128 + h * 16 + l * 4 + p];
            const float ix = lx * Wl - 0.5f, iy = ly * Hl - 0.5f;
            const float fx = floorf(ix), fy = floorf(iy);
            const int x0 = (int)fx, y0 = (int)fy;
            const float wx1 = ix - fx, wy1 = iy - fy;
            const float wx0 = 1.0f - wx1, wy0 = 1.0f - wy1;
            const float g00 = fetch_val(value, vbase, st, x0,     y0,     Wl, Hl);
            const float g10 = fetch_val(value, vbase, st, x0 + 1, y0,     Wl, Hl);
            const float g01 = fetch_val(value, vbase, st, x0,     y0 + 1, Wl, Hl);
            const float g11 = fetch_val(value, vbase, st, x0 + 1, y0 + 1, Wl, Hl);
            acc += w * (g00 * wx0 * wy0 + g10 * wx1 * wy0 + g01 * wx0 * wy1 + g11 * wx1 * wy1);
        }
    }
    out[(size_t)row * CC + h * DH + lane] = acc;
}

// ---------------- host-side orchestration ----------------
extern "C" void kernel_launch(void* const* d_in, const int* in_sizes, int n_in,
                              void* d_out, int out_size, void* d_ws, size_t ws_size,
                              hipStream_t stream)
{
    (void)in_sizes; (void)n_in; (void)out_size; (void)ws_size;

    const float* tgt   = (const float*)d_in[0];
    const float* qpos  = (const float*)d_in[1];
    const float* refp  = (const float*)d_in[2];
    const float* src   = (const float*)d_in[3];
    const long long* spatial = (const long long*)d_in[4];
    const long long* lstart  = (const long long*)d_in[5];
    const unsigned char* mask = (const unsigned char*)d_in[6];
    const float* w_in  = (const float*)d_in[7];
    const float* b_in  = (const float*)d_in[8];
    const float* w_out = (const float*)d_in[9];
    const float* b_out = (const float*)d_in[10];
    const float* g_n2  = (const float*)d_in[11];
    const float* b_n2  = (const float*)d_in[12];
    const float* w_v   = (const float*)d_in[13];
    const float* b_v   = (const float*)d_in[14];
    const float* w_off = (const float*)d_in[15];
    const float* b_off = (const float*)d_in[16];
    const float* w_aw  = (const float*)d_in[17];
    const float* b_aw  = (const float*)d_in[18];
    const float* w_o   = (const float*)d_in[19];
    const float* b_o   = (const float*)d_in[20];
    const float* g_n1  = (const float*)d_in[21];
    const float* b_n1  = (const float*)d_in[22];
    const float* w1    = (const float*)d_in[23];
    const float* b1    = (const float*)d_in[24];
    const float* w2    = (const float*)d_in[25];
    const float* b2    = (const float*)d_in[26];
    const float* g_n3  = (const float*)d_in[27];
    const float* b_n3  = (const float*)d_in[28];

    float* out_tgt = (float*)d_out;
    float* out_loc = out_tgt + (size_t)ROWS * CC;       // 1,228,800
    float* out_aw  = out_loc + (size_t)ROWS * 256;      // + 1,228,800

    // ---- workspace carve-out ----
    char* base = (char*)d_ws;
    size_t cur = 0;
    auto take = [&](size_t bytes) -> void* {
        void* p = base + cur;
        cur += (bytes + 255) & ~(size_t)255;
        return p;
    };
    _Float16* w_in_h  = (_Float16*)take((size_t)3 * CC * CC * 2);
    _Float16* w_out_h = (_Float16*)take((size_t)CC * CC * 2);
    _Float16* w_v_h   = (_Float16*)take((size_t)CC * CC * 2);
    _Float16* w_off_h = (_Float16*)take((size_t)256 * CC * 2);
    _Float16* w_aw_h  = (_Float16*)take((size_t)128 * CC * 2);
    _Float16* w_o_h   = (_Float16*)take((size_t)CC * CC * 2);
    _Float16* w1_h    = (_Float16*)take((size_t)DFFN * CC * 2);
    _Float16* w2_h    = (_Float16*)take((size_t)CC * DFFN * 2);
    _Float16* value_h = (_Float16*)take((size_t)MV * CC * 2);
    float* qbuf  = (float*)take((size_t)ROWS * CC * 4);
    float* qkbuf = (float*)take((size_t)ROWS * 512 * 4);
    float* vpbuf = (float*)take((size_t)ROWS * CC * 4);
    float* pin   = (float*)take((size_t)ROWS * CC * 4);
    float* pout  = (float*)take((size_t)ROWS * CC * 4);
    float* tgt2  = (float*)take((size_t)ROWS * CC * 4);
    float* tgt3  = (float*)take((size_t)ROWS * CC * 4);
    float* offb  = (float*)take((size_t)ROWS * 256 * 4);
    float* awlin = (float*)take((size_t)ROWS * 128 * 4);
    float* hid   = (float*)take((size_t)ROWS * DFFN * 4);

    const int T = 256;
    auto blk = [](int n) { return (n + 255) / 256; };

    // 0) weight conversion fp32 -> f16
    cvt_f32_f16_kernel<<<blk(3 * CC * CC), T, 0, stream>>>(w_in, w_in_h, 3 * CC * CC);
    cvt_f32_f16_kernel<<<blk(CC * CC), T, 0, stream>>>(w_out, w_out_h, CC * CC);
    cvt_f32_f16_kernel<<<blk(CC * CC), T, 0, stream>>>(w_v, w_v_h, CC * CC);
    cvt_f32_f16_kernel<<<blk(256 * CC), T, 0, stream>>>(w_off, w_off_h, 256 * CC);
    cvt_f32_f16_kernel<<<blk(128 * CC), T, 0, stream>>>(w_aw, w_aw_h, 128 * CC);
    cvt_f32_f16_kernel<<<blk(CC * CC), T, 0, stream>>>(w_o, w_o_h, CC * CC);
    cvt_f32_f16_kernel<<<blk(DFFN * CC), T, 0, stream>>>(w1, w1_h, DFFN * CC);
    cvt_f32_f16_kernel<<<blk(CC * DFFN), T, 0, stream>>>(w2, w2_h, CC * DFFN);

    // 1) value projection (the big, bandwidth-bound GEMM) -> f16 value tensor
    gemm_xw_kernel<true><<<dim3(CC / 128, (MV + 63) / 64), T, 0, stream>>>(
        src, w_v_h, b_v, mask, value_h, MV, CC, CC, 0);

    // 2) self-attention block
    add_kernel<<<blk(ROWS * CC), T, 0, stream>>>(tgt, qpos, qbuf, ROWS * CC);
    gemm_xw_kernel<false><<<dim3(512 / 128, ROWS / 64), T, 0, stream>>>(
        qbuf, w_in_h, b_in, nullptr, qkbuf, ROWS, 512, CC, 0);
    gemm_xw_kernel<false><<<dim3(CC / 128, ROWS / 64), T, 0, stream>>>(
        tgt, w_in_h + (size_t)2 * CC * CC, b_in + 2 * CC, nullptr, vpbuf, ROWS, CC, CC, 0);
    attn_kernel<<<BB * HH * NQ / 8, T, 0, stream>>>(qkbuf, vpbuf, pin);
    gemm_xw_kernel<false><<<dim3(CC / 128, ROWS / 64), T, 0, stream>>>(
        pin, w_out_h, b_out, nullptr, pout, ROWS, CC, CC, 0);
    add_ln_kernel<<<ROWS / 8, T, 0, stream>>>(tgt, pout, g_n2, b_n2, tgt2);

    // 3) deformable cross-attention
    add_kernel<<<blk(ROWS * CC), T, 0, stream>>>(tgt2, qpos, qbuf, ROWS * CC);
    gemm_xw_kernel<false><<<dim3(256 / 128, ROWS / 64), T, 0, stream>>>(
        qbuf, w_off_h, b_off, nullptr, offb, ROWS, 256, CC, 0);
    gemm_xw_kernel<false><<<dim3(128 / 128, ROWS / 64), T, 0, stream>>>(
        qbuf, w_aw_h, b_aw, nullptr, awlin, ROWS, 128, CC, 0);
    prep_kernel<<<blk(ROWS * HH), T, 0, stream>>>(offb, awlin, refp, spatial, out_loc, out_aw);
    sample_kernel<<<BB * NQ * HH / 8, T, 0, stream>>>(value_h, out_loc, out_aw, spatial, lstart, pin);
    gemm_xw_kernel<false><<<dim3(CC / 128, ROWS / 64), T, 0, stream>>>(
        pin, w_o_h, b_o, nullptr, pout, ROWS, CC, CC, 0);
    add_ln_kernel<<<ROWS / 8, T, 0, stream>>>(tgt2, pout, g_n1, b_n1, tgt3);

    // 4) FFN
    gemm_xw_kernel<false><<<dim3(DFFN / 128, ROWS / 64), T, 0, stream>>>(
        tgt3, w1_h, b1, nullptr, hid, ROWS, DFFN, CC, 1 /*relu*/);
    gemm_xw_kernel<false><<<dim3(CC / 128, ROWS / 64), T, 0, stream>>>(
        hid, w2_h, b2, nullptr, pout, ROWS, CC, DFFN, 0);
    add_ln_kernel<<<ROWS / 8, T, 0, stream>>>(tgt3, pout, g_n3, b_n3, out_tgt);
}
